// Model_78529182040156
// MI455X (gfx1250) — compile-verified, matrix operand order checked by reading, not run on hardware
//
#include <hip/hip_runtime.h>

// ---------------------------------------------------------------------------
// POYO-style perceiver forward for MI455X (gfx1250), bf16 WMMA everywhere.
// B=32, S=2048, D=512, L=256, NC=64, DH=64, CROSS_H=1, SELF_H=8
// GEMM A-tile staging via Tensor Data Mover (tensor_load_to_lds, 6-arg form
// + s_wait_tensorcnt), guarded by __has_builtin with a synchronous fallback.
// ---------------------------------------------------------------------------

#define BB 32
#define SS 2048
#define DDIM 512
#define LL 256
#define NCOUT 64

#define DEV __device__ __forceinline__

typedef __attribute__((ext_vector_type(16))) __bf16        v16bf;
typedef __attribute__((ext_vector_type(8)))  float         v8f;
typedef __attribute__((ext_vector_type(4)))  unsigned int  u32x4;

#if defined(__AMDGCN__) && __has_builtin(__builtin_amdgcn_tensor_load_to_lds) && \
    __has_builtin(__builtin_amdgcn_s_wait_tensorcnt)
#define USE_TDM 1
typedef __attribute__((ext_vector_type(4))) unsigned int tdmv4u;
typedef __attribute__((ext_vector_type(8))) int          tdmv8i;
typedef __attribute__((ext_vector_type(4))) int          tdmv4i;
#else
#define USE_TDM 0
#endif

union Frag { v16bf v; u32x4 q[2]; };

DEV unsigned short f2bf(float f) {
  unsigned int u = __float_as_uint(f);
  u += 0x7FFFu + ((u >> 16) & 1u);          // round-to-nearest-even
  return (unsigned short)(u >> 16);
}
DEV float bf2f(unsigned short h) { return __uint_as_float(((unsigned int)h) << 16); }

DEV v8f wmma_bf16(const Frag& a, const Frag& b, v8f c) {
  return __builtin_amdgcn_wmma_f32_16x16x32_bf16(false, a.v, false, b.v,
                                                 (short)0, c, false, false);
}

// rotary inverse period: (2*pi) / (T_MIN * (T_MAX/T_MIN)^(i/32)),
// T_MIN=1e-4, T_MAX=4.0 -> ratio 40000
DEV float rope_inv(int i) {
  return 6.283185307179586f / (1.0e-4f * powf(40000.0f, (float)i * (1.0f / 32.0f)));
}

// ---------------------------------------------------------------------------
// Elementwise helpers
// ---------------------------------------------------------------------------
__global__ void k_cast(const float* __restrict__ s, unsigned short* __restrict__ d, int n) {
  int g = blockIdx.x * blockDim.x + threadIdx.x;
  if (g < n) d[g] = f2bf(s[g]);
}

__global__ void k_bcast(const float* __restrict__ s, float* __restrict__ d,
                        int srcN, int total) {
  int g = blockIdx.x * blockDim.x + threadIdx.x;
  if (g < total) d[g] = s[g % srcN];
}

// ---------------------------------------------------------------------------
// Embedding gather + input projection (K=74) + enc ln_c, fused per token.
// One block (128 threads) per token; output bf16 [B*S, 512].
// ---------------------------------------------------------------------------
__global__ __launch_bounds__(128) void k_embed_ln(
    const int* __restrict__ sid, const int* __restrict__ sub, const int* __restrict__ cid,
    const float* __restrict__ prom, const float* __restrict__ dur,
    const float* __restrict__ sessT, const float* __restrict__ subjT,
    const float* __restrict__ chanT,
    const float* __restrict__ projW /*[74,512]*/, const float* __restrict__ projB,
    const float* __restrict__ lnG, const float* __restrict__ lnB,
    unsigned short* __restrict__ out)
{
  int tok = blockIdx.x;
  int t = threadIdx.x;
  __shared__ float sF[74];
  if (t < 32)       sF[t] = sessT[(size_t)sid[tok] * 32 + t];
  else if (t < 64)  sF[t] = subjT[(size_t)sub[tok] * 32 + (t - 32)];
  else if (t < 72)  sF[t] = chanT[(size_t)cid[tok] * 8 + (t - 64)];
  else if (t == 72) sF[72] = prom[tok];
  else if (t == 73) sF[73] = dur[tok];
  __syncthreads();

  float r[4]; float s = 0.f, s2 = 0.f;
  #pragma unroll
  for (int j = 0; j < 4; j++) {
    int c = t + j * 128;
    float acc = projB[c];
    for (int f = 0; f < 74; f++) acc += sF[f] * projW[f * 512 + c];
    r[j] = acc; s += acc; s2 += acc * acc;
  }
  __shared__ float red[128], red2[128];
  red[t] = s; red2[t] = s2; __syncthreads();
  for (int off = 64; off > 0; off >>= 1) {
    if (t < off) { red[t] += red[t + off]; red2[t] += red2[t + off]; }
    __syncthreads();
  }
  float mean = red[0] * (1.f / 512.f);
  float var  = red2[0] * (1.f / 512.f) - mean * mean;
  float rs   = rsqrtf(var + 1e-5f);
  #pragma unroll
  for (int j = 0; j < 4; j++) {
    int c = t + j * 128;
    out[(size_t)tok * 512 + c] = f2bf((r[j] - mean) * rs * lnG[c] + lnB[c]);
  }
}

// ---------------------------------------------------------------------------
// LayerNorm (D=512 rows, f32 in -> bf16 out), one block per row.
// ---------------------------------------------------------------------------
__global__ __launch_bounds__(128) void k_ln(
    const float* __restrict__ x, const float* __restrict__ g,
    const float* __restrict__ b, unsigned short* __restrict__ out)
{
  int row = blockIdx.x, t = threadIdx.x;
  const float* xr = x + (size_t)row * 512;
  float loc[4], s = 0.f, s2 = 0.f;
  #pragma unroll
  for (int j = 0; j < 4; j++) {
    float v = xr[t + j * 128]; loc[j] = v; s += v; s2 += v * v;
  }
  __shared__ float r1[128], r2[128];
  r1[t] = s; r2[t] = s2; __syncthreads();
  for (int off = 64; off > 0; off >>= 1) {
    if (t < off) { r1[t] += r1[t + off]; r2[t] += r2[t + off]; }
    __syncthreads();
  }
  float mean = r1[0] * (1.f / 512.f);
  float var  = r2[0] * (1.f / 512.f) - mean * mean;
  float rs   = rsqrtf(var + 1e-5f);
  #pragma unroll
  for (int j = 0; j < 4; j++) {
    int c = t + j * 128;
    out[(size_t)row * 512 + c] = f2bf((loc[j] - mean) * rs * g[c] + b[c]);
  }
}

// ---------------------------------------------------------------------------
// Generic bf16 x bf16 -> f32 GEMM, C = A[MxK] * B[KxN] (+bias) (+resid).
// 128 threads = 4 waves; wave grid 2x2, each wave 2x2 WMMA tiles -> 64x64 block.
// A tile (64x32) staged by the Tensor Data Mover when available; B tile staged
// transposed by scalar loads. K multiple of 32, N multiple of 64; M arbitrary.
// f32out: 0 -> Cb bf16, 1 -> Cf f32 (+resid if non-null).
// ---------------------------------------------------------------------------
__global__ __launch_bounds__(128) void k_gemm(
    const unsigned short* __restrict__ A, const unsigned short* __restrict__ Bw,
    unsigned short* __restrict__ Cb, float* __restrict__ Cf,
    const float* __restrict__ bias, const float* __restrict__ resid,
    int M, int N, int K, int f32out)
{
  __shared__ __align__(16) unsigned short sA[64][32];  // dense 64B rows (TDM dest)
  __shared__ __align__(16) unsigned short sB[64][40];  // transposed tile [n][k]

  const int tid  = threadIdx.x;
  const int lane = tid & 31;
  const int wave = tid >> 5;
  const int hi   = (lane >> 4) & 1;
  const int l15  = lane & 15;
  const int wm   = (wave >> 1) * 32;
  const int wn   = (wave & 1) * 32;
  const int m0   = blockIdx.y * 64;
  const int n0   = blockIdx.x * 64;

  const v8f vz = {0.f, 0.f, 0.f, 0.f, 0.f, 0.f, 0.f, 0.f};
  v8f acc[2][2] = {{vz, vz}, {vz, vz}};

  const int ar = tid >> 1;         // A row in tile (fallback path)
  const int ak = (tid & 1) * 16;   // A k-offset (fallback path)

  for (int kc = 0; kc < K; kc += 32) {
#if USE_TDM
    // ---- stage A tile (64 rows x 32 cols, bf16) via Tensor Data Mover ----
    if (wave == 0) {
      unsigned long long ga =
          (unsigned long long)(size_t)(A + (size_t)m0 * K + kc);
      unsigned int lds = (unsigned int)(size_t)&sA[0][0];
      tdmv4u g0;
      g0.x = 1u;                                    // count=1, user descriptor
      g0.y = lds;                                   // lds_addr (bytes)
      g0.z = (unsigned int)ga;                      // global_addr[31:0]
      g0.w = (unsigned int)((ga >> 32) & 0x01FFFFFFu) | (2u << 30);  // addr hi | type=2
      int td0 = K - kc;                             // remaining tensor extent dim0
      int td1 = M - m0;                             // remaining rows (OOB -> zero fill)
      tdmv8i g1;
      g1[0] = 0x10000;                              // workgroup_mask=0, data_size=1 (2B)
      g1[1] = (td0 & 0xFFFF) << 16;                 // tensor_dim0[15:0] @ bits 63:48
      g1[2] = ((td0 >> 16) & 0xFFFF) | ((td1 & 0xFFFF) << 16);  // dim0 hi | dim1 lo
      g1[3] = ((td1 >> 16) & 0xFFFF) | (32 << 16);  // dim1 hi | tile_dim0 = 32
      g1[4] = 64;                                   // tile_dim1 = 64, tile_dim2 = 0
      g1[5] = K;                                    // tensor_dim0_stride[31:0]
      g1[6] = 0;                                    // stride hi | dim1_stride lo
      g1[7] = 0;
      tdmv4i gz4 = {0, 0, 0, 0};
      tdmv8i gz8 = {0, 0, 0, 0, 0, 0, 0, 0};
      __builtin_amdgcn_tensor_load_to_lds(g0, g1, gz4, gz4, gz8, 0);
    }
#else
    // ---- stage A tile (64x32) synchronously ----
    {
      u32x4 z = {0u, 0u, 0u, 0u};
      u32x4 d0 = z, d1 = z;
      int gm = m0 + ar;
      if (gm < M) {
        const unsigned short* p = A + (size_t)gm * K + kc + ak;
        d0 = *(const u32x4*)p;
        d1 = *(const u32x4*)(p + 8);
        if (kc + 32 < K) __builtin_prefetch(p + 32, 0, 1);
      }
      *(u32x4*)&sA[ar][ak]     = d0;
      *(u32x4*)&sA[ar][ak + 8] = d1;
    }
#endif
    // ---- stage B tile transposed: sB[n][k] = B[kc+k][n0+n] ----
    #pragma unroll
    for (int i = 0; i < 16; i++) {
      int idx = tid + 128 * i;
      int n = idx & 63, k = idx >> 6;
      sB[n][k] = Bw[(size_t)(kc + k) * N + n0 + n];
    }
#if USE_TDM
    __builtin_amdgcn_s_wait_tensorcnt(0);   // A tile landed in LDS (all waves: cheap)
#endif
    __syncthreads();

    Frag a0, a1, b0, b1;
    {
      int r = wm + l15;
      a0.q[0] = *(const u32x4*)&sA[r][hi * 8];
      a0.q[1] = *(const u32x4*)&sA[r][16 + hi * 8];
      r = wm + 16 + l15;
      a1.q[0] = *(const u32x4*)&sA[r][hi * 8];
      a1.q[1] = *(const u32x4*)&sA[r][16 + hi * 8];
      int c = wn + l15;
      b0.q[0] = *(const u32x4*)&sB[c][hi * 8];
      b0.q[1] = *(const u32x4*)&sB[c][16 + hi * 8];
      c = wn + 16 + l15;
      b1.q[0] = *(const u32x4*)&sB[c][hi * 8];
      b1.q[1] = *(const u32x4*)&sB[c][16 + hi * 8];
    }
    acc[0][0] = wmma_bf16(a0, b0, acc[0][0]);
    acc[0][1] = wmma_bf16(a0, b1, acc[0][1]);
    acc[1][0] = wmma_bf16(a1, b0, acc[1][0]);
    acc[1][1] = wmma_bf16(a1, b1, acc[1][1]);
    __syncthreads();
  }

  // ---- epilogue ----
  #pragma unroll
  for (int i = 0; i < 2; i++) {
    #pragma unroll
    for (int j = 0; j < 2; j++) {
      #pragma unroll
      for (int p = 0; p < 8; p++) {
        int row = m0 + wm + i * 16 + p + 8 * hi;
        int col = n0 + wn + j * 16 + l15;
        if (row < M) {
          float v = acc[i][j][p];
          if (bias) v += bias[col];
          size_t idx = (size_t)row * N + col;
          if (f32out) {
            float rr = resid ? resid[idx] : 0.f;
            Cf[idx] = rr + v;
          } else {
            Cb[idx] = f2bf(v);
          }
        }
      }
    }
  }
}

// ---------------------------------------------------------------------------
// Single-wave flash attention with WMMA. Grid: (ceil(Nq/16), B*H).
// Q [bh, Nq, 64] bf16, K [bh, Nk, 64] bf16, VT [bh, 64, Nk] bf16 (pre-transposed),
// O [bh, Nq, 64] f32. seqlen (per batch) masks keys >= seqlen[b] (or null).
// ---------------------------------------------------------------------------
__global__ __launch_bounds__(32) void k_attn(
    const unsigned short* __restrict__ Q, const unsigned short* __restrict__ Kmat,
    const unsigned short* __restrict__ VT, float* __restrict__ O,
    const int* __restrict__ seqlen, int H, int Nq, int Nk)
{
  const int lane = threadIdx.x;
  const int hi = lane >> 4, l15 = lane & 15;
  const int q0 = blockIdx.x * 16;
  const int bh = blockIdx.y;
  const int b  = bh / H;
  const unsigned short* Qb = Q    + (size_t)bh * Nq * 64;
  const unsigned short* Kb = Kmat + (size_t)bh * Nk * 64;
  const unsigned short* Vb = VT   + (size_t)bh * 64 * Nk;
  float* Ob = O + (size_t)bh * Nq * 64;

  __shared__ __align__(16) float          sS[16][32];
  __shared__ __align__(16) unsigned short sP[16][32];
  __shared__ float sM[16], sL[16], sAl[16];
  if (lane < 16) { sM[lane] = -3.0e38f; sL[lane] = 0.f; }

  // Q fragments (dh chunks 0..31, 32..63), read directly in WMMA A striping
  Frag aq[2];
  {
    int qr = q0 + l15; if (qr >= Nq) qr = Nq - 1;
    const unsigned short* qp = Qb + (size_t)qr * 64;
    #pragma unroll
    for (int c = 0; c < 2; c++) {
      aq[c].q[0] = *(const u32x4*)(qp + c * 32 + hi * 8);
      aq[c].q[1] = *(const u32x4*)(qp + c * 32 + 16 + hi * 8);
    }
  }
  const v8f vz = {0.f, 0.f, 0.f, 0.f, 0.f, 0.f, 0.f, 0.f};
  v8f ao[4] = {vz, vz, vz, vz};
  const int slen = seqlen ? seqlen[b] : Nk;

  for (int kb = 0; kb < Nk; kb += 32) {
    // ---- S = Q * K^T for 32 keys (two 16x16 tiles) ----
    v8f s0 = vz, s1 = vz;
    #pragma unroll
    for (int c = 0; c < 2; c++) {
      Frag bk;
      const unsigned short* kp = Kb + (size_t)(kb + l15) * 64 + c * 32;
      bk.q[0] = *(const u32x4*)(kp + hi * 8);
      bk.q[1] = *(const u32x4*)(kp + 16 + hi * 8);
      s0 = wmma_bf16(aq[c], bk, s0);
      const unsigned short* kp2 = Kb + (size_t)(kb + 16 + l15) * 64 + c * 32;
      bk.q[0] = *(const u32x4*)(kp2 + hi * 8);
      bk.q[1] = *(const u32x4*)(kp2 + 16 + hi * 8);
      s1 = wmma_bf16(aq[c], bk, s1);
    }
    __syncthreads();  // previous iteration's sP/sAl reads done before overwrite
    #pragma unroll
    for (int p = 0; p < 8; p++) {
      int row = p + 8 * hi;
      float v0 = s0[p] * 0.125f;                 // SCALE = 1/sqrt(64)
      if (kb + l15 >= slen) v0 = -1e30f;
      sS[row][l15] = v0;
      float v1 = s1[p] * 0.125f;
      if (kb + 16 + l15 >= slen) v1 = -1e30f;
      sS[row][16 + l15] = v1;
    }
    __syncthreads();
    // ---- online softmax row stats (lanes 0..15, one row each) ----
    if (lane < 16) {
      float mo = sM[lane], mn = mo;
      #pragma unroll
      for (int j = 0; j < 32; j++) mn = fmaxf(mn, sS[lane][j]);
      float al = expf(mo - mn);
      float ls = sL[lane] * al;
      #pragma unroll
      for (int j = 0; j < 32; j++) {
        float e = expf(sS[lane][j] - mn);
        sP[lane][j] = f2bf(e);
        ls += e;
      }
      sM[lane] = mn; sL[lane] = ls; sAl[lane] = al;
    }
    __syncthreads();
    // ---- O = O*alpha + P * V (P re-striped via LDS, V^T from global) ----
    Frag ap;
    ap.q[0] = *(const u32x4*)&sP[l15][hi * 8];
    ap.q[1] = *(const u32x4*)&sP[l15][16 + hi * 8];
    #pragma unroll
    for (int t = 0; t < 4; t++) {
      #pragma unroll
      for (int p = 0; p < 8; p++) ao[t][p] *= sAl[p + 8 * hi];
      Frag bv;
      const unsigned short* vp = Vb + (size_t)(t * 16 + l15) * Nk + kb;
      bv.q[0] = *(const u32x4*)(vp + hi * 8);
      bv.q[1] = *(const u32x4*)(vp + 16 + hi * 8);
      ao[t] = wmma_bf16(ap, bv, ao[t]);
    }
  }
  __syncthreads();
  #pragma unroll
  for (int t = 0; t < 4; t++) {
    #pragma unroll
    for (int p = 0; p < 8; p++) {
      int row = p + 8 * hi;
      int qr = q0 + row;
      if (qr < Nq) Ob[(size_t)qr * 64 + t * 16 + l15] = ao[t][p] / sL[row];
    }
  }
}

// ---------------------------------------------------------------------------
// Rotary kernels
// ---------------------------------------------------------------------------
__global__ void k_rope_q(const unsigned short* __restrict__ in,
                         unsigned short* __restrict__ out,
                         const float* __restrict__ ts, int Lsz, int rows, int tsConst)
{
  int g = blockIdx.x * blockDim.x + threadIdx.x;
  if (g >= rows * 32) return;
  int i = g & 31, row = g >> 5;
  float t = tsConst ? 1.0f : ts[row % Lsz];
  float a = t * rope_inv(i);
  float c = cosf(a), sn = sinf(a);
  float x1 = bf2f(in[(size_t)row * 64 + i]);
  float x2 = bf2f(in[(size_t)row * 64 + 32 + i]);
  out[(size_t)row * 64 + i]      = f2bf(x1 * c - x2 * sn);
  out[(size_t)row * 64 + 32 + i] = f2bf(x2 * c + x1 * sn);
}

// encoder context: kv [B*S,128] -> rotated K [B*S,64], rotated V^T [B,64,S]
__global__ void k_rope_enc_kv(const unsigned short* __restrict__ kv,
                              const float* __restrict__ ts,
                              unsigned short* __restrict__ Ko,
                              unsigned short* __restrict__ VTo, int Stot)
{
  int g = blockIdx.x * blockDim.x + threadIdx.x;
  if (g >= BB * Stot * 32) return;
  int i = g & 31, tok = g >> 5;
  int b = tok / Stot, s = tok % Stot;
  float a = ts[tok] * rope_inv(i);
  float c = cosf(a), sn = sinf(a);
  const unsigned short* r = kv + (size_t)tok * 128;
  float k1 = bf2f(r[i]),      k2 = bf2f(r[32 + i]);
  float v1 = bf2f(r[64 + i]), v2 = bf2f(r[96 + i]);
  Ko[(size_t)tok * 64 + i]      = f2bf(k1 * c - k2 * sn);
  Ko[(size_t)tok * 64 + 32 + i] = f2bf(k2 * c + k1 * sn);
  VTo[((size_t)b * 64 + i) * Stot + s]      = f2bf(v1 * c - v2 * sn);
  VTo[((size_t)b * 64 + 32 + i) * Stot + s] = f2bf(v2 * c + v1 * sn);
}

// self-attn: qkv [B*L,1536] (8 heads) -> rotated Q,K [B,8,L,64], rotated V^T [B,8,64,L]
__global__ void k_rope_self_qkv(const unsigned short* __restrict__ qkv,
                                const float* __restrict__ ts,
                                unsigned short* __restrict__ Qo,
                                unsigned short* __restrict__ Ko,
                                unsigned short* __restrict__ VTo, int Lsz)
{
  int g = blockIdx.x * blockDim.x + threadIdx.x;
  if (g >= BB * Lsz * 8 * 32) return;
  int i = g & 31;
  int h = (g >> 5) & 7;
  int tok = g >> 8;
  int b = tok / Lsz, l = tok % Lsz;
  float a = ts[l] * rope_inv(i);
  float c = cosf(a), sn = sinf(a);
  const unsigned short* r = qkv + (size_t)tok * 1536;
  float q1 = bf2f(r[h * 64 + i]),        q2 = bf2f(r[h * 64 + 32 + i]);
  float k1 = bf2f(r[512 + h * 64 + i]),  k2 = bf2f(r[512 + h * 64 + 32 + i]);
  float v1 = bf2f(r[1024 + h * 64 + i]), v2 = bf2f(r[1024 + h * 64 + 32 + i]);
  size_t base = (((size_t)b * 8 + h) * Lsz + l) * 64;
  Qo[base + i]      = f2bf(q1 * c - q2 * sn);
  Qo[base + 32 + i] = f2bf(q2 * c + q1 * sn);
  Ko[base + i]      = f2bf(k1 * c - k2 * sn);
  Ko[base + 32 + i] = f2bf(k2 * c + k1 * sn);
  size_t vtb = ((size_t)b * 8 + h) * 64;
  VTo[(vtb + i) * Lsz + l]      = f2bf(v1 * c - v2 * sn);
  VTo[(vtb + 32 + i) * Lsz + l] = f2bf(v2 * c + v1 * sn);
}

// decoder: kv [B*L,128] -> rotated K [B*L,64], plain V^T [B,64,L] (rotate_value=False)
__global__ void k_rope_dec_kv(const unsigned short* __restrict__ kv,
                              const float* __restrict__ ts,
                              unsigned short* __restrict__ Ko,
                              unsigned short* __restrict__ VTo, int Lsz)
{
  int g = blockIdx.x * blockDim.x + threadIdx.x;
  if (g >= BB * Lsz * 32) return;
  int i = g & 31, tok = g >> 5;
  int b = tok / Lsz, l = tok % Lsz;
  float a = ts[l] * rope_inv(i);
  float c = cosf(a), sn = sinf(a);
  const unsigned short* r = kv + (size_t)tok * 128;
  float k1 = bf2f(r[i]), k2 = bf2f(r[32 + i]);
  Ko[(size_t)tok * 64 + i]      = f2bf(k1 * c - k2 * sn);
  Ko[(size_t)tok * 64 + 32 + i] = f2bf(k2 * c + k1 * sn);
  VTo[((size_t)b * 64 + i) * Lsz + l]      = r[64 + i];
  VTo[((size_t)b * 64 + 32 + i) * Lsz + l] = r[96 + i];
}

// un-rotate attention output with -f_query and merge heads: O f32 [B,H,N,64] -> bf16 [B,N,H*64]
__global__ void k_unrot_merge(const float* __restrict__ O, unsigned short* __restrict__ out,
                              const float* __restrict__ ts, int H, int N, int rot, int total)
{
  int g = blockIdx.x * blockDim.x + threadIdx.x;
  if (g >= total) return;
  int i = g & 31;
  int rest = g >> 5;
  int n = rest % N; rest /= N;
  int h = rest % H; int b = rest / H;
  const float* op = O + (((size_t)b * H + h) * N + n) * 64;
  float o1 = op[i], o2 = op[32 + i];
  float r1 = o1, r2 = o2;
  if (rot) {
    float a = -ts[n] * rope_inv(i);
    float c = cosf(a), sn = sinf(a);
    r1 = o1 * c - o2 * sn;
    r2 = o2 * c + o1 * sn;
  }
  size_t ob = ((size_t)b * N + n) * ((size_t)H * 64) + (size_t)h * 64;
  out[ob + i]      = f2bf(r1);
  out[ob + 32 + i] = f2bf(r2);
}

// GEGLU: in [rows, W] -> out [rows, W/2];  out = a * gelu_tanh(g)
__global__ void k_geglu(const unsigned short* __restrict__ in,
                        unsigned short* __restrict__ out, int total, int W)
{
  int g = blockIdx.x * blockDim.x + threadIdx.x;
  if (g >= total) return;
  int halfW = W >> 1;
  int row = g / halfW, j = g - row * halfW;
  float a = bf2f(in[(size_t)row * W + j]);
  float x = bf2f(in[(size_t)row * W + halfW + j]);
  float t = tanhf(0.7978845608028654f * (x + 0.044715f * x * x * x));
  out[(size_t)row * halfW + j] = f2bf(a * 0.5f * x * (1.f + t));
}

// ---------------------------------------------------------------------------
// Parameter leaf indices (depth-first, insertion order) offset by 9 inputs.
// ---------------------------------------------------------------------------
enum {
  P_SESS = 0, P_SUBJ, P_CHAN, P_LAT_TABLE, P_PROJW, P_PROJB, P_CLSQ,
  P_ENC_LNQG, P_ENC_LNQB, P_ENC_LNCG, P_ENC_LNCB, P_ENC_WQ, P_ENC_WKV, P_ENC_WO, P_ENC_BO,
  P_EFF_LNG, P_EFF_LNB, P_EFF_W1, P_EFF_B1, P_EFF_W2, P_EFF_B2,
  P_P0A_LNG, P_P0A_LNB, P_P0A_WQKV, P_P0A_WO, P_P0A_BO,
  P_P0F_LNG, P_P0F_LNB, P_P0F_W1, P_P0F_B1, P_P0F_W2, P_P0F_B2,
  P_P1A_LNG, P_P1A_LNB, P_P1A_WQKV, P_P1A_WO, P_P1A_BO,
  P_P1F_LNG, P_P1F_LNB, P_P1F_W1, P_P1F_B1, P_P1F_W2, P_P1F_B2,
  P_DEC_LNQG, P_DEC_LNQB, P_DEC_LNCG, P_DEC_LNCB, P_DEC_WQ, P_DEC_WKV, P_DEC_WO, P_DEC_BO,
  P_DFF_LNG, P_DFF_LNB, P_DFF_W1, P_DFF_B1, P_DFF_W2, P_DFF_B2,
  P_LNF_G, P_LNF_B, P_RD_W, P_RD_B
};

extern "C" void kernel_launch(void* const* d_in, const int* in_sizes, int n_in,
                              void* d_out, int out_size, void* d_ws, size_t ws_size,
                              hipStream_t stream)
{
  (void)in_sizes; (void)n_in; (void)out_size; (void)ws_size;

  const int*   sid     = (const int*)d_in[0];
  const int*   sub     = (const int*)d_in[1];
  const int*   cid     = (const int*)d_in[2];
  const float* prom    = (const float*)d_in[3];
  const float* dur     = (const float*)d_in[4];
  const int*   seqlen  = (const int*)d_in[5];
  const float* tstamps = (const float*)d_in[6];
  const float* lat_ts  = (const float*)d_in[8];   // d_in[7] latent_idx == arange(L)
  auto PP = [&](int i) { return (const float*)d_in[9 + i]; };

  // ---- deterministic bump allocator over d_ws ----
  char* wsb = (char*)d_ws;
  size_t cur = 0;
  auto alloc = [&](size_t bytes) -> void* {
    cur = (cur + 255) & ~(size_t)255;
    void* p = wsb + cur; cur += bytes; return p;
  };
  auto allocBF = [&](size_t e) { return (unsigned short*)alloc(e * 2); };
  auto allocF  = [&](size_t e) { return (float*)alloc(e * 4); };

  auto castw = [&](int pi, size_t n) {
    unsigned short* d = allocBF(n);
    k_cast<<<(unsigned)((n + 255) / 256), 256, 0, stream>>>(PP(pi), d, (int)n);
    return d;
  };

  // ---- weights -> bf16 ----
  unsigned short* wEncQ  = castw(P_ENC_WQ,  512 * 64);
  unsigned short* wEncKV = castw(P_ENC_WKV, 512 * 128);
  unsigned short* wEncO  = castw(P_ENC_WO,  64 * 512);
  unsigned short* wW1[4] = { castw(P_EFF_W1, 512 * 4096), castw(P_P0F_W1, 512 * 4096),
                             castw(P_P1F_W1, 512 * 4096), castw(P_DFF_W1, 512 * 4096) };
  unsigned short* wW2[4] = { castw(P_EFF_W2, 2048 * 512), castw(P_P0F_W2, 2048 * 512),
                             castw(P_P1F_W2, 2048 * 512), castw(P_DFF_W2, 2048 * 512) };
  unsigned short* wQkv[2] = { castw(P_P0A_WQKV, 512 * 1536), castw(P_P1A_WQKV, 512 * 1536) };
  unsigned short* wSWo[2] = { castw(P_P0A_WO, 512 * 512),   castw(P_P1A_WO, 512 * 512) };
  unsigned short* wDecQ  = castw(P_DEC_WQ,  512 * 64);
  unsigned short* wDecKV = castw(P_DEC_WKV, 512 * 128);
  unsigned short* wDecO  = castw(P_DEC_WO,  64 * 512);
  unsigned short* wRd    = castw(P_RD_W,    512 * 64);

  // ---- activation buffers ----
  float*          lat    = allocF((size_t)BB * LL * DDIM);          // latent residual stream
  float*          oq     = allocF((size_t)BB * DDIM);               // output-query stream
  unsigned short* arenaA = allocBF((size_t)BB * SS * DDIM);         // lnx / ffn hidden
  unsigned short* arenaB = allocBF((size_t)BB * LL * 2048);         // kv / qkv / geglu
  unsigned short* Kenc   = allocBF((size_t)BB * SS * 64);
  unsigned short* VTenc  = allocBF((size_t)BB * 64 * SS);
  unsigned short* lnlat  = allocBF((size_t)BB * LL * DDIM);
  unsigned short* qbuf   = allocBF((size_t)BB * LL * 64);
  unsigned short* qrot   = allocBF((size_t)BB * LL * 64);
  float*          Obuf   = allocF((size_t)BB * 8 * LL * 64);
  unsigned short* merged = allocBF((size_t)BB * LL * 512);
  unsigned short* sQ     = allocBF((size_t)BB * 8 * LL * 64);
  unsigned short* sK     = allocBF((size_t)BB * 8 * LL * 64);
  unsigned short* sVT    = allocBF((size_t)BB * 8 * 64 * LL);
  unsigned short* decK   = allocBF((size_t)BB * LL * 64);
  unsigned short* decVT  = allocBF((size_t)BB * 64 * LL);
  unsigned short* lnoq   = allocBF((size_t)BB * 512);
  unsigned short* qd     = allocBF((size_t)BB * 64);
  unsigned short* qdr    = allocBF((size_t)BB * 64);
  float*          Od     = allocF((size_t)BB * 64);
  unsigned short* Odb    = allocBF((size_t)BB * 64);
  unsigned short* lnf    = allocBF((size_t)BB * 512);

  auto gemm = [&](const unsigned short* A, const unsigned short* W, void* C,
                  const float* bias, const float* resid, int M, int N, int K, int f32out) {
    dim3 grd((unsigned)(N / 64), (unsigned)((M + 63) / 64));
    k_gemm<<<grd, 128, 0, stream>>>(A, W,
        f32out ? nullptr : (unsigned short*)C, f32out ? (float*)C : nullptr,
        bias, resid, M, N, K, f32out);
  };

  auto ffn = [&](float* state, int rows, int lnGi, int lnBi,
                 unsigned short* w1, int b1i, unsigned short* w2, int b2i,
                 unsigned short* lnbuf) {
    k_ln<<<(unsigned)rows, 128, 0, stream>>>(state, PP(lnGi), PP(lnBi), lnbuf);
    gemm(lnbuf, w1, arenaA, PP(b1i), nullptr, rows, 4096, 512, 0);
    int tot = rows * 2048;
    k_geglu<<<(unsigned)((tot + 255) / 256), 256, 0, stream>>>(arenaA, arenaB, tot, 4096);
    gemm(arenaB, w2, state, PP(b2i), state, rows, 512, 2048, 1);
  };

  // ================= init =================
  {
    int totLat = BB * LL * DDIM;
    k_bcast<<<(unsigned)((totLat + 255) / 256), 256, 0, stream>>>(PP(P_LAT_TABLE), lat, LL * DDIM, totLat);
    k_bcast<<<(unsigned)((BB * DDIM + 255) / 256), 256, 0, stream>>>(PP(P_CLSQ), oq, DDIM, BB * DDIM);
  }

  // ================= embed + enc ln_c =================
  k_embed_ln<<<(unsigned)(BB * SS), 128, 0, stream>>>(
      sid, sub, cid, prom, dur,
      PP(P_SESS), PP(P_SUBJ), PP(P_CHAN), PP(P_PROJW), PP(P_PROJB),
      PP(P_ENC_LNCG), PP(P_ENC_LNCB), arenaA);

  // ================= encoder cross-attention =================
  gemm(arenaA, wEncKV, arenaB, nullptr, nullptr, BB * SS, 128, 512, 0);          // kv
  k_ln<<<(unsigned)(BB * LL), 128, 0, stream>>>(lat, PP(P_ENC_LNQG), PP(P_ENC_LNQB), lnlat);
  gemm(lnlat, wEncQ, qbuf, nullptr, nullptr, BB * LL, 64, 512, 0);               // q
  k_rope_q<<<(unsigned)((BB * LL * 32 + 255) / 256), 256, 0, stream>>>(qbuf, qrot, lat_ts, LL, BB * LL, 0);
  k_rope_enc_kv<<<(unsigned)((BB * SS * 32 + 255) / 256), 256, 0, stream>>>(arenaB, tstamps, Kenc, VTenc, SS);
  k_attn<<<dim3(LL / 16, BB), 32, 0, stream>>>(qrot, Kenc, VTenc, Obuf, seqlen, 1, LL, SS);
  k_unrot_merge<<<(unsigned)((BB * LL * 32 + 255) / 256), 256, 0, stream>>>(
      Obuf, merged, lat_ts, 1, LL, 1, BB * 1 * LL * 32);
  gemm(merged, wEncO, lat, PP(P_ENC_BO), lat, BB * LL, 512, 64, 1);              // + residual
  ffn(lat, BB * LL, P_EFF_LNG, P_EFF_LNB, wW1[0], P_EFF_B1, wW2[0], P_EFF_B2, lnlat);

  // ================= 2 x rotary self-attention blocks =================
  for (int bi = 0; bi < 2; bi++) {
    int A0 = P_P0A_LNG + bi * 11;
    int F0 = P_P0F_LNG + bi * 11;
    k_ln<<<(unsigned)(BB * LL), 128, 0, stream>>>(lat, PP(A0), PP(A0 + 1), lnlat);
    gemm(lnlat, wQkv[bi], arenaB, nullptr, nullptr, BB * LL, 1536, 512, 0);      // qkv
    k_rope_self_qkv<<<(unsigned)((BB * LL * 8 * 32 + 255) / 256), 256, 0, stream>>>(
        arenaB, lat_ts, sQ, sK, sVT, LL);
    k_attn<<<dim3(LL / 16, BB * 8), 32, 0, stream>>>(sQ, sK, sVT, Obuf, nullptr, 8, LL, LL);
    k_unrot_merge<<<(unsigned)((BB * 8 * LL * 32 + 255) / 256), 256, 0, stream>>>(
        Obuf, merged, lat_ts, 8, LL, 1, BB * 8 * LL * 32);
    gemm(merged, wSWo[bi], lat, PP(A0 + 4), lat, BB * LL, 512, 512, 1);
    ffn(lat, BB * LL, F0, F0 + 1, wW1[1 + bi], F0 + 3, wW2[1 + bi], F0 + 5, lnlat);
  }

  // ================= decoder cross-attention =================
  k_ln<<<(unsigned)BB, 128, 0, stream>>>(oq, PP(P_DEC_LNQG), PP(P_DEC_LNQB), lnoq);
  gemm(lnoq, wDecQ, qd, nullptr, nullptr, BB, 64, 512, 0);
  k_rope_q<<<(unsigned)((BB * 32 + 255) / 256), 256, 0, stream>>>(qd, qdr, nullptr, 1, BB, 1);
  k_ln<<<(unsigned)(BB * LL), 128, 0, stream>>>(lat, PP(P_DEC_LNCG), PP(P_DEC_LNCB), lnlat);
  gemm(lnlat, wDecKV, arenaB, nullptr, nullptr, BB * LL, 128, 512, 0);
  k_rope_dec_kv<<<(unsigned)((BB * LL * 32 + 255) / 256), 256, 0, stream>>>(arenaB, lat_ts, decK, decVT, LL);
  k_attn<<<dim3(1, BB), 32, 0, stream>>>(qdr, decK, decVT, Od, nullptr, 1, 1, LL);
  k_unrot_merge<<<(unsigned)((BB * 32 + 255) / 256), 256, 0, stream>>>(
      Od, Odb, lat_ts, 1, 1, 0, BB * 32);
  gemm(Odb, wDecO, oq, PP(P_DEC_BO), oq, BB, 512, 64, 1);
  ffn(oq, BB, P_DFF_LNG, P_DFF_LNB, wW1[3], P_DFF_B1, wW2[3], P_DFF_B2, lnoq);

  // ================= final LN + readout =================
  k_ln<<<(unsigned)BB, 128, 0, stream>>>(oq, PP(P_LNF_G), PP(P_LNF_B), lnf);
  gemm(lnf, wRd, (float*)d_out, PP(P_RD_B), nullptr, BB, 64, 512, 1);
}
